// PointNet_58858231824472
// MI455X (gfx1250) — compile-verified, matrix operand order checked by reading, not run on hardware
//
#include <hip/hip_runtime.h>

typedef __attribute__((ext_vector_type(16))) _Float16 v16h;
typedef __attribute__((ext_vector_type(8)))  _Float16 v8h;
typedef __attribute__((ext_vector_type(8)))  float    v8f;

union V16 { v16h v; v8h h[2]; };

#define HDIM 128

// ---------------------------------------------------------------------------
// Pre-swizzle a weight matrix B[128,Nc] (fp32, row-major) into f16 WMMA
// B-fragment layout: out[((tn*4 + kb)*32 + lane)*16 + e], where the (lane,e)
// mapping matches v_wmma_f32_16x16x32_f16's B operand:
//   col = tn*16 + (lane&15),  k = kb*32 + (lane>>4)*8 + (e&7) + (e>>3)*16
// Columns >= Nc are zero-padded (handles OUT_C=40 with 3 column tiles).
// ---------------------------------------------------------------------------
__global__ void swizzle_weights(const float* __restrict__ B,
                                _Float16* __restrict__ out,
                                int Nc, int total)
{
    int t = blockIdx.x * blockDim.x + threadIdx.x;
    if (t >= total) return;
    int e    = t & 15;
    int lane = (t >> 4) & 31;
    int kb   = (t >> 9) & 3;
    int tn   = t >> 11;
    int k    = kb * 32 + (lane >> 4) * 8 + (e & 7) + ((e >> 3) << 4);
    int col  = tn * 16 + (lane & 15);
    float v  = (col < Nc) ? B[(size_t)k * Nc + col] : 0.0f;
    out[t] = (_Float16)v;   // layout index == t by construction
}

// ---------------------------------------------------------------------------
// WMMA GEMM: C[M,Nc] = act(A[M,128] @ B[128,Nc] + bias), optional second
// output C2 = A@B + bias2 (seeds the self-loop max-aggregation).
// One block = one 16-row tile; 256 threads stage the 16x128 A tile into LDS
// as f16 once, then each wave32 computes one 16x16 output tile with 4
// v_wmma_f32_16x16x32_f16 ops. Per K-step: 2x ds_load_b128 (A) + one 32B
// global load of the pre-swizzled B fragment + 1 WMMA. fp32 accumulate.
// ---------------------------------------------------------------------------
__global__ __launch_bounds__(256)
void wmma_gemm_k128(const float* __restrict__ A,
                    const _Float16* __restrict__ Bsw,   // [tilesN][4][32][16] f16
                    const float* __restrict__ bias,
                    float* __restrict__ C,
                    float* __restrict__ C2,
                    const float* __restrict__ bias2,
                    int M, int tilesN, int Nc, int doRelu)
{
    __shared__ _Float16 Ash[16 * HDIM];   // 4 KB
    const int tm  = blockIdx.x;
    const int tid = threadIdx.x;

    // cooperative A-tile stage: 2048 fp32 -> f16, 8 elements per thread
    {
        int idx = tid * 8;                    // [0, 2048)
        int row = idx >> 7;
        int gr  = tm * 16 + row;
        if (gr >= M) gr = M - 1;              // clamp; stores masked below
        const float* src = A + (size_t)gr * HDIM + (idx & 127);
        float4 p0 = *(const float4*)(src);
        float4 p1 = *(const float4*)(src + 4);
        v8h h;
        h[0] = (_Float16)p0.x; h[1] = (_Float16)p0.y;
        h[2] = (_Float16)p0.z; h[3] = (_Float16)p0.w;
        h[4] = (_Float16)p1.x; h[5] = (_Float16)p1.y;
        h[6] = (_Float16)p1.z; h[7] = (_Float16)p1.w;
        *(v8h*)(&Ash[idx]) = h;
    }
    __syncthreads();

    const int wv   = tid >> 5;                // wave id == column tile tn
    const int lane = tid & 31;
    if (wv >= tilesN) return;                 // wave-uniform (EXEC stays full)
    const int half = lane >> 4;
    const int l15  = lane & 15;

    v8f acc = {};
    const _Float16* Bp   = Bsw + ((size_t)wv * 128 + lane) * 16;
    const _Float16* Arow = &Ash[l15 * HDIM];
#pragma unroll
    for (int kb = 0; kb < 4; ++kb) {
        int k0 = kb * 32 + half * 8;
        V16 a, b;
        a.h[0] = *(const v8h*)(Arow + k0);
        a.h[1] = *(const v8h*)(Arow + k0 + 16);
        b.v    = *(const v16h*)(Bp + (size_t)kb * 512);
        acc = __builtin_amdgcn_wmma_f32_16x16x32_f16(
                  false, a.v, false, b.v, (short)0, acc, false, false);
    }

    int bcol = wv * 16 + l15;
    if (bcol >= Nc) return;                   // divergence only after WMMA
    float bv  = bias  ? bias[bcol]  : 0.0f;
    float bv2 = bias2 ? bias2[bcol] : 0.0f;
    int mbase = tm * 16 + half * 8;
#pragma unroll
    for (int r = 0; r < 8; ++r) {
        int m = mbase + r;
        if (m >= M) continue;
        float v = acc[r] + bv;
        C[(size_t)m * Nc + bcol] = doRelu ? fmaxf(v, 0.0f) : v;
        if (C2) C2[(size_t)m * Nc + bcol] = acc[r] + bv2;
    }
}

// pos = h @ Wp2[128,3] + bp2 : one wave per node, shuffle reduction.
__global__ void pos_kernel(const float* __restrict__ h,
                           const float* __restrict__ Wp2,
                           const float* __restrict__ bp2,
                           float* __restrict__ pos, int n)
{
    int wid  = blockIdx.x * (blockDim.x >> 5) + (threadIdx.x >> 5);
    int lane = threadIdx.x & 31;
    if (wid >= n) return;
    const float4 v = *(const float4*)(h + (size_t)wid * HDIM + lane * 4);
    float s0 = 0.f, s1 = 0.f, s2 = 0.f;
    const float* vp = &v.x;
#pragma unroll
    for (int j = 0; j < 4; ++j) {
        int k = lane * 4 + j;
        float xv = vp[j];
        s0 += xv * Wp2[k * 3 + 0];
        s1 += xv * Wp2[k * 3 + 1];
        s2 += xv * Wp2[k * 3 + 2];
    }
    for (int off = 16; off > 0; off >>= 1) {
        s0 += __shfl_down(s0, off, 32);
        s1 += __shfl_down(s1, off, 32);
        s2 += __shfl_down(s2, off, 32);
    }
    if (lane == 0) {
        pos[3 * (size_t)wid + 0] = s0 + bp2[0];
        pos[3 * (size_t)wid + 1] = s1 + bp2[1];
        pos[3 * (size_t)wid + 2] = s2 + bp2[2];
    }
}

// rel[e] = pos[src[e]] - pos[dst[e]]  (loop-invariant, computed once)
__global__ void rel_kernel(const float* __restrict__ pos,
                           const int* __restrict__ src,
                           const int* __restrict__ dst,
                           float* __restrict__ rel, int nE)
{
    int e = blockIdx.x * blockDim.x + threadIdx.x;
    if (e >= nE) return;
    int j = src[e], i = dst[e];
    rel[3 * (size_t)e + 0] = pos[3 * (size_t)j + 0] - pos[3 * (size_t)i + 0];
    rel[3 * (size_t)e + 1] = pos[3 * (size_t)j + 1] - pos[3 * (size_t)i + 1];
    rel[3 * (size_t)e + 2] = pos[3 * (size_t)j + 2] - pos[3 * (size_t)i + 2];
}

// float atomic max via signed-max (v>=0) / unsigned-min (v<0) bit tricks
__device__ __forceinline__ void atomicMaxF(float* addr, float v)
{
    if (v >= 0.0f) atomicMax((int*)addr, __float_as_int(v));
    else           atomicMin((unsigned int*)addr, __float_as_uint(v));
}

// One wave per edge: msg = y[src] + bl + rel @ Wl3 ; agg[dst] = max(agg, msg)
__global__ void edge_msg_kernel(const float* __restrict__ y,
                                const float* __restrict__ rel,
                                const int* __restrict__ src,
                                const int* __restrict__ dst,
                                const float* __restrict__ Wl3,  // [3][128]
                                const float* __restrict__ bl,
                                float* __restrict__ agg, int nE)
{
    int wid  = blockIdx.x * (blockDim.x >> 5) + (threadIdx.x >> 5);
    int lane = threadIdx.x & 31;
    if (wid >= nE) return;
    int j = src[wid];
    int i = dst[wid];
    float rx = rel[3 * (size_t)wid + 0];
    float ry = rel[3 * (size_t)wid + 1];
    float rz = rel[3 * (size_t)wid + 2];
    int c = lane * 4;
    const float4 v  = *(const float4*)(y + (size_t)j * HDIM + c);
    const float4 w0 = *(const float4*)(Wl3 + c);
    const float4 w1 = *(const float4*)(Wl3 + HDIM + c);
    const float4 w2 = *(const float4*)(Wl3 + 2 * HDIM + c);
    const float4 bb = *(const float4*)(bl + c);
    float m0 = v.x + bb.x + rx * w0.x + ry * w1.x + rz * w2.x;
    float m1 = v.y + bb.y + rx * w0.y + ry * w1.y + rz * w2.y;
    float m2 = v.z + bb.z + rx * w0.z + ry * w1.z + rz * w2.z;
    float m3 = v.w + bb.w + rx * w0.w + ry * w1.w + rz * w2.w;
    float* ap = agg + (size_t)i * HDIM + c;
    atomicMaxF(ap + 0, m0);
    atomicMaxF(ap + 1, m1);
    atomicMaxF(ap + 2, m2);
    atomicMaxF(ap + 3, m3);
}

extern "C" void kernel_launch(void* const* d_in, const int* in_sizes, int n_in,
                              void* d_out, int out_size, void* d_ws, size_t ws_size,
                              hipStream_t stream)
{
    (void)n_in; (void)out_size; (void)ws_size;
    const float* x    = (const float*)d_in[0];
    const int*   edge = (const int*)  d_in[1];
    const float* W1   = (const float*)d_in[2];  const float* b1  = (const float*)d_in[3];
    const float* W2   = (const float*)d_in[4];  const float* b2  = (const float*)d_in[5];
    const float* Wp1  = (const float*)d_in[6];  const float* bp1 = (const float*)d_in[7];
    const float* Wp2  = (const float*)d_in[8];  const float* bp2 = (const float*)d_in[9];
    const float* Wl   = (const float*)d_in[10]; const float* bl  = (const float*)d_in[11];
    const float* Wg   = (const float*)d_in[12]; const float* bg  = (const float*)d_in[13];
    const float* Wf   = (const float*)d_in[14]; const float* bf  = (const float*)d_in[15];

    const int N    = in_sizes[0] / HDIM;
    const int E    = in_sizes[1] / 2;
    const int OUTC = in_sizes[15];
    const int* src = edge;        // edge_index[0] = source j
    const int* dst = edge + E;    // edge_index[1] = target i

    // workspace layout (fp32 regions, then f16 swizzled weights) ~174 MB
    size_t nH   = (size_t)N * HDIM;
    float* tmp  = (float*)d_ws;          // h1 / p1 / per-layer y
    float* xbuf = tmp  + nH;             // node features x
    float* agg  = xbuf + nH;             // max-aggregation accumulator
    float* pos  = agg  + nH;             // [N,3]
    float* rel  = pos  + (size_t)N * 3;  // [E,3]
    _Float16* wsw = (_Float16*)(rel + (size_t)E * 3);
    const int tilesNo = (OUTC + 15) / 16;
    const int T8 = 8 * 2048;             // halves per 8-col-tile matrix
    _Float16* sW1  = wsw;
    _Float16* sW2  = sW1 + T8;
    _Float16* sWp1 = sW2 + T8;
    _Float16* sWlx = sWp1 + T8;
    _Float16* sWg  = sWlx + T8;
    _Float16* sWf  = sWg + T8;           // tilesNo*2048 halves

    const int tilesM = (N + 15) / 16;
    const dim3 blk(256);

    // swizzle all weight matrices to f16 fragment layout (tiny, per-launch)
    swizzle_weights<<<dim3(64), blk, 0, stream>>>(W1,  sW1,  HDIM, T8);
    swizzle_weights<<<dim3(64), blk, 0, stream>>>(W2,  sW2,  HDIM, T8);
    swizzle_weights<<<dim3(64), blk, 0, stream>>>(Wp1, sWp1, HDIM, T8);
    swizzle_weights<<<dim3(64), blk, 0, stream>>>(Wl,  sWlx, HDIM, T8);  // first 128 rows
    swizzle_weights<<<dim3(64), blk, 0, stream>>>(Wg,  sWg,  HDIM, T8);
    swizzle_weights<<<dim3((unsigned)(tilesNo * 8)), blk, 0, stream>>>(
        Wf, sWf, OUTC, tilesNo * 2048);

    const dim3 ggrid((unsigned)tilesM);
    // initial MLP: tmp = relu(x@W1+b1); xbuf = tmp@W2+b2
    wmma_gemm_k128<<<ggrid, blk, 0, stream>>>(x,   sW1, b1,  tmp,  nullptr, nullptr, N, 8, HDIM, 1);
    wmma_gemm_k128<<<ggrid, blk, 0, stream>>>(tmp, sW2, b2,  xbuf, nullptr, nullptr, N, 8, HDIM, 0);
    // pos_gen: tmp = relu(xbuf@Wp1+bp1); pos = tmp@Wp2+bp2
    wmma_gemm_k128<<<ggrid, blk, 0, stream>>>(xbuf, sWp1, bp1, tmp, nullptr, nullptr, N, 8, HDIM, 1);
    pos_kernel<<<dim3((unsigned)((N + 7) / 8)), blk, 0, stream>>>(tmp, Wp2, bp2, pos, N);
    rel_kernel<<<dim3((unsigned)((E + 255) / 256)), blk, 0, stream>>>(pos, src, dst, rel, E);

    const float* Wl3 = Wl + (size_t)HDIM * HDIM;  // last 3 rows of Wl
    for (int l = 0; l < 3; ++l) {
        // y = x @ Wl[:128] ; agg seeded with self-loop message y + bl
        wmma_gemm_k128<<<ggrid, blk, 0, stream>>>(xbuf, sWlx, nullptr, tmp, agg, bl, N, 8, HDIM, 0);
        // scatter-max messages over real edges
        edge_msg_kernel<<<dim3((unsigned)((E + 7) / 8)), blk, 0, stream>>>(
            tmp, rel, src, dst, Wl3, bl, agg, E);
        // x = relu(agg @ Wg + bg)
        wmma_gemm_k128<<<ggrid, blk, 0, stream>>>(agg, sWg, bg, xbuf, nullptr, nullptr, N, 8, HDIM, 1);
    }

    // final linear: out = x @ Wf + bf  (Nc=40, zero-padded column tiles)
    wmma_gemm_k128<<<ggrid, blk, 0, stream>>>(
        xbuf, sWf, bf, (float*)d_out, nullptr, nullptr, N, tilesNo, OUTC, 0);
}
